// GATLayerMultiStream_71768903516799
// MI455X (gfx1250) — matrix operand order repaired and checked by reading, not compiled, not run on hardware
//
#include <hip/hip_runtime.h>

// MI455X / gfx1250, wave32. All matrix math via v_wmma_f32_16x16x32_bf16.

typedef __attribute__((ext_vector_type(16))) __bf16 v16bf;
typedef __attribute__((ext_vector_type(8)))  float  v8f;

union V16 { v16bf v; unsigned short u[16]; };

__device__ __forceinline__ unsigned short f2bf(float f) {
  union { float f; unsigned u; } x; x.f = f;
  unsigned u = x.u;
  return (unsigned short)((u + 0x7FFFu + ((u >> 16) & 1u)) >> 16);  // RNE
}
__device__ __forceinline__ float bf2f(unsigned short s) {
  union { unsigned u; float f; } x; x.u = ((unsigned)s) << 16; return x.f;
}
__device__ __forceinline__ float elu_f(float x)   { return x > 0.f ? x : (expf(x) - 1.f); }
__device__ __forceinline__ float lrelu_f(float x) { return x >= 0.f ? x : 0.2f * x; }

#define NSITE 8192   // 32 * 16 * 16 sites (b,h,w)

// ---- workspace layout (bytes) ----
static const size_t WT_OFF    = 0;                    // W_map^T bf16      [256][256]  131072
static const size_t CF_OFF    = 131072;               // coeff^T bf16      [16][256]     8192
static const size_t ADJ_OFF   = 139264;               // adj_n^T f32       [16][16]      1024
static const size_t WHT_OFF   = 140288;               // Wh^T bf16  [8192][256][16]  67108864
static const size_t SU_OFF    = 67249152;             // s1,s2,u1,u2 f32 [4][8192][16] 2097152
static const size_t ATTSP_OFF = 69346304;             // att_sp f32 [8192][16][16]    8388608
static const size_t ATTT_OFF  = 77734912;             // att_t  f32 [8192][16][16]    8388608

// ================= k0: prep W_map^T(bf16), coeff^T(bf16), adj_n^T =================
__global__ __launch_bounds__(256) void k0_prep(const float* __restrict__ Wmap,
    const float* __restrict__ asp, const float* __restrict__ atm,
    const float* __restrict__ Bm,
    unsigned short* __restrict__ Wt, unsigned short* __restrict__ coefT,
    float* __restrict__ adjT) {
  int tid = threadIdx.x;
  for (int i = tid; i < 256 * 256; i += 256) {
    int n = i >> 8, k = i & 255;
    Wt[n * 256 + k] = f2bf(Wmap[k * 256 + n]);
  }
  for (int i = tid; i < 16 * 256; i += 256) {
    int q = i >> 8, t = i & 255;
    float v = 0.f;
    if (q == 0) v = asp[t]; else if (q == 1) v = asp[256 + t];
    else if (q == 2) v = atm[t]; else if (q == 3) v = atm[256 + t];
    coefT[i] = f2bf(v);
  }
  __shared__ float am[256];
  __shared__ float red[256];
  __shared__ float dinv[16];
  int i = tid >> 4, j = tid & 15;
  float av = Bm[tid] + (i == j ? 1.f : 0.f);
  red[tid] = av; __syncthreads();
  for (int s = 128; s > 0; s >>= 1) { if (tid < s) red[tid] = fminf(red[tid], red[tid + s]); __syncthreads(); }
  float mn = red[0]; __syncthreads();
  red[tid] = av; __syncthreads();
  for (int s = 128; s > 0; s >>= 1) { if (tid < s) red[tid] = fmaxf(red[tid], red[tid + s]); __syncthreads(); }
  float mx = red[0];
  float an = (av - mn) / (mx - mn);
  am[tid] = an; __syncthreads();
  if (tid < 16) {
    float s = 0.f;
    #pragma unroll
    for (int jj = 0; jj < 16; ++jj) s += am[tid * 16 + jj];
    dinv[tid] = rsqrtf(s);
  }
  __syncthreads();
  adjT[j * 16 + i] = an * dinv[i] * dinv[j];   // adjT[k][i] = adj_n[i][k]
}

// ============ k1: Wh^T[site][t][v] = (h_site^T @ W_map), bf16 WMMA, K=256 ============
__global__ __launch_bounds__(256) void k1_wh(const float* __restrict__ h,
    const unsigned short* __restrict__ Wt, unsigned short* __restrict__ WhT) {
  __shared__ unsigned short A_lds[16][272];            // [v][t] bf16 (padded for 16B alignment)
  int site = blockIdx.x;
  const float* hp = h + (size_t)site * 256 * 16;       // h[site][t][v]
  int tid = threadIdx.x;
  {
    int t = tid;                                       // transpose on the fly
    const float* row = hp + t * 16;
    #pragma unroll
    for (int v = 0; v < 16; ++v) A_lds[v][t] = f2bf(row[v]);
  }
  __syncthreads();
  int lane = tid & 31, wave = tid >> 5;
  int col = lane & 15, hi = lane >> 4;
  int akb = hi ? 8 : 0, bkb = hi ? 16 : 0, m0 = hi * 8;
  int n0 = wave * 32 + col;                            // out cols (t) for the two tiles
  int n1 = n0 + 16;
  v8f acc0 = {}, acc1 = {};
  #pragma unroll
  for (int kc = 0; kc < 8; ++kc) {
    int t0 = kc * 32;
    V16 a;
    #pragma unroll
    for (int e = 0; e < 8; ++e) {
      a.u[e]     = A_lds[col][t0 + bkb + e];
      a.u[8 + e] = A_lds[col][t0 + bkb + 8 + e];
    }
    const unsigned short* w0 = Wt + n0 * 256 + t0 + bkb;
    const unsigned short* w1 = Wt + n1 * 256 + t0 + bkb;
    if (kc < 7) __builtin_prefetch(w0 + 32, 0, 1);     // global_prefetch next K chunk
    V16 b0, b1;
    #pragma unroll
    for (int e = 0; e < 16; ++e) { b0.u[e] = w0[e]; b1.u[e] = w1[e]; }
    acc0 = __builtin_amdgcn_wmma_f32_16x16x32_bf16(false, a.v, false, b0.v, (short)0, acc0, false, false);
    acc1 = __builtin_amdgcn_wmma_f32_16x16x32_bf16(false, a.v, false, b1.v, (short)0, acc1, false, false);
  }
  unsigned short* o0 = WhT + ((size_t)site * 256 + n0) * 16 + m0;
  unsigned short* o1 = WhT + ((size_t)site * 256 + n1) * 16 + m0;
  #pragma unroll
  for (int r = 0; r < 8; ++r) { o0[r] = f2bf(acc0[r]); o1[r] = f2bf(acc1[r]); }
}

// ========== k2: s1/s2/u1/u2 as a 16x256 @ 256x16 WMMA vs coeff matrix ==========
__global__ __launch_bounds__(256) void k2_su(const unsigned short* __restrict__ WhT,
    const unsigned short* __restrict__ coefT, float* __restrict__ su) {
  int tid = threadIdx.x;
  int lane = tid & 31, wave = tid >> 5;
  int site = blockIdx.x * 8 + wave;                    // one wave per site
  int col = lane & 15, hi = lane >> 4;
  int akb = hi ? 8 : 0, bkb = hi ? 16 : 0, m0 = hi * 8;
  const unsigned short* wp = WhT + (size_t)site * 4096; // [t][v]
  v8f acc = {};
  #pragma unroll
  for (int kc = 0; kc < 8; ++kc) {
    int t0 = kc * 32;
    V16 a, b;
    #pragma unroll
    for (int e = 0; e < 8; ++e) {
      a.u[e]     = wp[(t0 + bkb + e) * 16 + col];      // A[v=col][k=t]
      a.u[8 + e] = wp[(t0 + bkb + 8 + e) * 16 + col];
    }
    const unsigned short* cp = coefT + col * 256 + t0 + bkb;
    #pragma unroll
    for (int e = 0; e < 16; ++e) b.u[e] = cp[e];
    acc = __builtin_amdgcn_wmma_f32_16x16x32_bf16(false, a.v, false, b.v, (short)0, acc, false, false);
  }
  if (col < 4) {                                       // cols 0..3 = s1,s2,u1,u2
    float* op = su + ((size_t)col * NSITE + site) * 16 + m0;
    #pragma unroll
    for (int r = 0; r < 8; ++r) op[r] = acc[r];
  }
}

// ================= k3: softmax over w for att_sp and att_t =================
__global__ __launch_bounds__(256) void k3_att(const float* __restrict__ su,
    float* __restrict__ att_sp, float* __restrict__ att_t) {
  const float* s1 = su;
  const float* s2 = su + (size_t)NSITE * 16;
  const float* u1 = su + (size_t)2 * NSITE * 16;
  const float* u2 = su + (size_t)3 * NSITE * 16;
  __shared__ float S1[16][16], S2[16][16], U1[16][16], U2[16][16];  // [w][v]
  int site0 = blockIdx.x * 16;                         // (b,h) block; +w gives site
  int tid = threadIdx.x;
  int w = tid >> 4, v = tid & 15;
  S1[w][v] = s1[(site0 + w) * 16 + v];
  S2[w][v] = s2[(site0 + w) * 16 + v];
  U1[w][v] = u1[(site0 + w) * 16 + v];
  U2[w][v] = u2[(site0 + w) * 16 + v];
  __syncthreads();
  int i = tid >> 4, j = tid & 15;
  float z[16], mx, sum, inv;
  mx = -3.4e38f;
  #pragma unroll
  for (int ww = 0; ww < 16; ++ww) { float x = lrelu_f(S1[ww][i] + S2[ww][j]); z[ww] = x; mx = fmaxf(mx, x); }
  sum = 0.f;
  #pragma unroll
  for (int ww = 0; ww < 16; ++ww) { float e = expf(z[ww] - mx); z[ww] = e; sum += e; }
  inv = 1.f / sum;
  #pragma unroll
  for (int ww = 0; ww < 16; ++ww) att_sp[(size_t)(site0 + ww) * 256 + i * 16 + j] = z[ww] * inv;
  mx = -3.4e38f;
  #pragma unroll
  for (int ww = 0; ww < 16; ++ww) { float x = lrelu_f(U1[ww][i] + U2[ww][j]); z[ww] = x; mx = fmaxf(mx, x); }
  sum = 0.f;
  #pragma unroll
  for (int ww = 0; ww < 16; ++ww) { float e = expf(z[ww] - mx); z[ww] = e; sum += e; }
  inv = 1.f / sum;
  #pragma unroll
  for (int ww = 0; ww < 16; ++ww) att_t[(size_t)(site0 + ww) * 256 + i * 16 + j] = z[ww] * inv;
}

// ====== k4: hs branch. D2 = adj_n^T @ (att_sp @ Wh_site), elu, scattered store ======
__global__ __launch_bounds__(256) void k4_hs(const unsigned short* __restrict__ WhT,
    const float* __restrict__ att_sp, const float* __restrict__ adjT,
    float* __restrict__ out1) {
  __shared__ unsigned short H1[256][16];               // hs1^T[t][i] bf16
  int site = blockIdx.x;
  int b = site >> 8, h = (site >> 4) & 15, w = site & 15;
  int tid = threadIdx.x, lane = tid & 31, wave = tid >> 5;
  int col = lane & 15, hi = lane >> 4, m0 = hi * 8, akb = hi ? 8 : 0;
  V16 a1;                                              // A = att_sp (K=16, padded to 32)
  {
    const float* ap = att_sp + (size_t)site * 256 + col * 16;
    #pragma unroll
    for (int e = 0; e < 16; ++e) a1.u[e] = hi ? (unsigned short)0 : f2bf(ap[e]);
  }
  #pragma unroll
  for (int p = 0; p < 2; ++p) {
    int t = (wave * 2 + p) * 16 + col;
    V16 bm;
    if (!hi) {
      const unsigned short* bp = WhT + ((size_t)site * 256 + t) * 16;  // B[k=j][n=t]
      #pragma unroll
      for (int e = 0; e < 16; ++e) bm.u[e] = bp[e];
    } else {
      #pragma unroll
      for (int e = 0; e < 16; ++e) bm.u[e] = 0;        // padded K half
    }
    v8f c = {};
    c = __builtin_amdgcn_wmma_f32_16x16x32_bf16(false, a1.v, false, bm.v, (short)0, c, false, false);
    #pragma unroll
    for (int r = 0; r < 8; ++r) H1[t][m0 + r] = f2bf(c[r]);
  }
  __syncthreads();
  V16 a2;                                              // A = adj_n^T
  {
    const float* ap = adjT + col * 16;
    #pragma unroll
    for (int e = 0; e < 16; ++e) a2.u[e] = hi ? (unsigned short)0 : f2bf(ap[e]);
  }
  #pragma unroll
  for (int p = 0; p < 2; ++p) {
    int t = (wave * 2 + p) * 16 + col;
    V16 bm;
    if (!hi) {
      #pragma unroll
      for (int e = 0; e < 16; ++e) bm.u[e] = H1[t][e];
    } else {
      #pragma unroll
      for (int e = 0; e < 16; ++e) bm.u[e] = 0;
    }
    v8f c = {};
    c = __builtin_amdgcn_wmma_f32_16x16x32_bf16(false, a2.v, false, bm.v, (short)0, c, false, false);
    // out1 flat layout: (b, w, t, h, k)
    float* op = out1 + ((((size_t)(b * 16 + w) * 256 + t) * 16 + h) * 16 + m0);
    #pragma unroll
    for (int r = 0; r < 8; ++r) op[r] = elu_f(c[r]);
  }
}

// ====== k5: ht branch. per (b,t): D2 = adj_n^T @ (att_t @ Wh[b,:,s,t]), elu ======
__global__ __launch_bounds__(256) void k5_ht(const unsigned short* __restrict__ WhT,
    const float* __restrict__ att_t, const float* __restrict__ adjT,
    float* __restrict__ out2) {
  __shared__ unsigned short BL[256][16];               // Wh slice [s][j] bf16
  __shared__ unsigned short H1[256][16];               // ht1^T[s][i] bf16
  int bt = blockIdx.x;
  int b = bt >> 8, t = bt & 255;
  int tid = threadIdx.x;
  {
    int s = tid;                                       // gather fixed-t slice (L2-resident)
    const unsigned short* p = WhT + ((size_t)(b * 256 + s) * 256 + t) * 16;
    #pragma unroll
    for (int j = 0; j < 16; ++j) BL[s][j] = p[j];
  }
  __syncthreads();
  int lane = tid & 31, wave = tid >> 5;
  int col = lane & 15, hi = lane >> 4, m0 = hi * 8, akb = hi ? 8 : 0;
  V16 a1;                                              // A = att_t at tmap site (= b*256+t)
  {
    const float* ap = att_t + (size_t)(b * 256 + t) * 256 + col * 16;
    #pragma unroll
    for (int e = 0; e < 16; ++e) a1.u[e] = hi ? (unsigned short)0 : f2bf(ap[e]);
  }
  #pragma unroll
  for (int p = 0; p < 2; ++p) {
    int s = (wave * 2 + p) * 16 + col;
    V16 bm;
    if (!hi) {
      #pragma unroll
      for (int e = 0; e < 16; ++e) bm.u[e] = BL[s][e];
    } else {
      #pragma unroll
      for (int e = 0; e < 16; ++e) bm.u[e] = 0;
    }
    v8f c = {};
    c = __builtin_amdgcn_wmma_f32_16x16x32_bf16(false, a1.v, false, bm.v, (short)0, c, false, false);
    #pragma unroll
    for (int r = 0; r < 8; ++r) H1[s][m0 + r] = f2bf(c[r]);
  }
  __syncthreads();
  V16 a2;
  {
    const float* ap = adjT + col * 16;
    #pragma unroll
    for (int e = 0; e < 16; ++e) a2.u[e] = hi ? (unsigned short)0 : f2bf(ap[e]);
  }
  #pragma unroll
  for (int p = 0; p < 2; ++p) {
    int s = (wave * 2 + p) * 16 + col;
    V16 bm;
    if (!hi) {
      #pragma unroll
      for (int e = 0; e < 16; ++e) bm.u[e] = H1[s][e];
    } else {
      #pragma unroll
      for (int e = 0; e < 16; ++e) bm.u[e] = 0;
    }
    v8f c = {};
    c = __builtin_amdgcn_wmma_f32_16x16x32_bf16(false, a2.v, false, bm.v, (short)0, c, false, false);
    // out2 flat layout: (b, h, w, t, k) with s = h*16+w
    float* op = out2 + (((size_t)(b * 256 + s) * 256 + t) * 16 + m0);
    #pragma unroll
    for (int r = 0; r < 8; ++r) op[r] = elu_f(c[r]);
  }
}

extern "C" void kernel_launch(void* const* d_in, const int* in_sizes, int n_in,
                              void* d_out, int out_size, void* d_ws, size_t ws_size,
                              hipStream_t stream) {
  (void)in_sizes; (void)n_in; (void)out_size; (void)ws_size;
  const float* h    = (const float*)d_in[0];
  const float* Wmap = (const float*)d_in[1];
  const float* asp  = (const float*)d_in[2];
  const float* atm  = (const float*)d_in[3];
  const float* Bm   = (const float*)d_in[4];
  char* ws = (char*)d_ws;
  unsigned short* Wt    = (unsigned short*)(ws + WT_OFF);
  unsigned short* coefT = (unsigned short*)(ws + CF_OFF);
  float*          adjT  = (float*)(ws + ADJ_OFF);
  unsigned short* WhT   = (unsigned short*)(ws + WHT_OFF);
  float*          su    = (float*)(ws + SU_OFF);
  float*          attsp = (float*)(ws + ATTSP_OFF);
  float*          attt  = (float*)(ws + ATTT_OFF);
  float* out1 = (float*)d_out;
  float* out2 = out1 + (size_t)32 * 16 * 16 * 256 * 16;

  k0_prep<<<1,    256, 0, stream>>>(Wmap, asp, atm, Bm, Wt, coefT, adjT);
  k1_wh  <<<8192, 256, 0, stream>>>(h, Wt, WhT);
  k2_su  <<<1024, 256, 0, stream>>>(WhT, coefT, su);
  k3_att <<<512,  256, 0, stream>>>(su, attsp, attt);
  k4_hs  <<<8192, 256, 0, stream>>>(WhT, attsp, adjT, out1);
  k5_ht  <<<8192, 256, 0, stream>>>(WhT, attt, adjT, out2);
}
